// AttnDecoderRNN_48215302865290
// MI455X (gfx1250) — matrix operand order repaired and checked by reading, not compiled, not run on hardware
//
#include <hip/hip_runtime.h>
#include <math.h>
#include <stdint.h>

// Sizes from the reference
#define V_SZ 50257
#define H_SZ 1024
#define L_SZ 2048
#define NL_SZ 5
#define MAX_K 2048   // largest GEMV depth (comb layer)

typedef __attribute__((ext_vector_type(16))) __bf16 v16bf;
typedef __attribute__((ext_vector_type(8)))  float  v8f;

// ---------------------------------------------------------------------------
// Generic GEMV via WMMA: out[r] = act( dot(W[r,:], v) + bias[r] )
// W: rows x K row-major f32, K % 32 == 0, K <= MAX_K.
//
// Data movement plan (MI455X):
//  * weights: streamed ONCE from HBM with global_load_b128 (binding resource,
//    roofline ~15us for the whole step at 23.3 TB/s) + global_prefetch_b8
//  * x-vector: staged into LDS once per workgroup via the gfx1250 async path
//    (global_load_async_to_lds_b128 / ASYNCcnt / s_wait_asynccnt), then read
//    with ds_load_b128 so the VMEM pipe stays dedicated to the weight stream
//  * MACs: v_wmma_f32_16x16x32_bf16, one wave per 16 output rows; x is
//    replicated across all 16 B-columns so every D column holds the same 16
//    dot products (bf16 keeps the f32 exponent; C accumulates in f32)
// ---------------------------------------------------------------------------
__global__ void gemv_wmma_bf16(const float* __restrict__ W,
                               const float* __restrict__ vvec,
                               const float* __restrict__ bias,
                               float* __restrict__ out,
                               int rows, int K, int relu) {
  __shared__ float xs[MAX_K];

  // --- Stage x into LDS with async global->LDS copies (16B per lane) ------
  // 256 threads x 4 floats = 1024 floats per round.
  for (int i = threadIdx.x * 4; i < K; i += 1024) {
    uint32_t lds_off = (uint32_t)(uintptr_t)(&xs[i]);   // low 32b = LDS offset
    const float* gp = vvec + i;
    asm volatile("global_load_async_to_lds_b128 %0, %1, off"
                 :: "v"(lds_off), "v"(gp)
                 : "memory");
  }
  asm volatile("s_wait_asynccnt 0" ::: "memory");
  __syncthreads();                                      // all waves arrive

  const int wave = threadIdx.x >> 5;                    // 8 waves / block
  const int lane = threadIdx.x & 31;
  const int r0   = (blockIdx.x * 8 + wave) * 16;        // first row of wave
  if (r0 >= rows) return;                               // wave-uniform exit

  const int m    = lane & 15;
  const int half = lane >> 4;                           // 0 or 1
  int row = r0 + m;
  if (row >= rows) row = rows - 1;                      // clamp, EXEC stays full
  const float* __restrict__ wrow = W + (size_t)row * (size_t)K;

  v8f acc = {};                                         // 16x16 f32 C/D

  for (int kb = 0; kb < K; kb += 32) {
    // Prefetch the weight stream ahead (global_prefetch_b8).
    __builtin_prefetch(wrow + kb + 256, 0, 1);

    // A (16x32 bf16): lane(m,half=0) holds K=kb+0..7 and kb+16..23,
    //                 lane(m,half=1) holds K=kb+8..15 and kb+24..31.
    const float4* p0 = (const float4*)(wrow + kb + 8 * half);
    const float4* p1 = (const float4*)(wrow + kb + 16 + 8 * half);
    float4 a0 = p0[0], a1 = p0[1];
    float4 a2 = p1[0], a3 = p1[1];
    v16bf A;
    A[0]  = (__bf16)a0.x; A[1]  = (__bf16)a0.y; A[2]  = (__bf16)a0.z; A[3]  = (__bf16)a0.w;
    A[4]  = (__bf16)a1.x; A[5]  = (__bf16)a1.y; A[6]  = (__bf16)a1.z; A[7]  = (__bf16)a1.w;
    A[8]  = (__bf16)a2.x; A[9]  = (__bf16)a2.y; A[10] = (__bf16)a2.z; A[11] = (__bf16)a2.w;
    A[12] = (__bf16)a3.x; A[13] = (__bf16)a3.y; A[14] = (__bf16)a3.z; A[15] = (__bf16)a3.w;

    // B (32x16 bf16) from LDS: lanes 0-15 hold K=kb+0..7 and kb+16..23, lanes 16-31
    // K=kb+8..15 and kb+24..31; identical for every column N (x replicated).
    const float4* q0 = (const float4*)(&xs[kb + 8 * half]);
    const float4* q1 = (const float4*)(&xs[kb + 16 + 8 * half]);
    float4 b0 = q0[0], b1 = q0[1], b2 = q1[0], b3 = q1[1];
    v16bf B;
    B[0]  = (__bf16)b0.x; B[1]  = (__bf16)b0.y; B[2]  = (__bf16)b0.z; B[3]  = (__bf16)b0.w;
    B[4]  = (__bf16)b1.x; B[5]  = (__bf16)b1.y; B[6]  = (__bf16)b1.z; B[7]  = (__bf16)b1.w;
    B[8]  = (__bf16)b2.x; B[9]  = (__bf16)b2.y; B[10] = (__bf16)b2.z; B[11] = (__bf16)b2.w;
    B[12] = (__bf16)b3.x; B[13] = (__bf16)b3.y; B[14] = (__bf16)b3.z; B[15] = (__bf16)b3.w;

    acc = __builtin_amdgcn_wmma_f32_16x16x32_bf16(
        /*neg_a=*/false, A, /*neg_b=*/false, B,
        /*c_mod=*/(short)0, acc, /*reuse_a=*/false, /*reuse_b=*/false);
  }

  // D: lanes 0-15 hold M=0..7 (VGPR g -> M=g), lanes 16-31 hold M=8..15.
  // All columns equal; lane 0 writes rows r0..r0+7, lane 16 writes r0+8..+15.
  if (m == 0) {
    const int base = r0 + 8 * half;
    #pragma unroll
    for (int g = 0; g < 8; ++g) {
      const int rr = base + g;
      if (rr < rows) {
        float val = acc[g] + (bias ? bias[rr] : 0.0f);
        if (relu) val = fmaxf(val, 0.0f);
        out[rr] = val;
      }
    }
  }
}

// ---------------------------------------------------------------------------
// Softmax over L=2048 scores -> weights (ws copy + d_out copy).
// ---------------------------------------------------------------------------
__global__ void softmax2048(const float* __restrict__ scores,
                            float* __restrict__ w_ws,
                            float* __restrict__ w_out) {
  __shared__ float red[256];
  const int t = threadIdx.x;
  float m = -INFINITY;
  for (int i = t; i < L_SZ; i += 256) m = fmaxf(m, scores[i]);
  red[t] = m; __syncthreads();
  for (int s = 128; s > 0; s >>= 1) { if (t < s) red[t] = fmaxf(red[t], red[t + s]); __syncthreads(); }
  const float mx = red[0];
  __syncthreads();
  float sum = 0.0f;
  for (int i = t; i < L_SZ; i += 256) {
    float e = expf(scores[i] - mx);
    w_ws[i] = e;                                    // stash unnormalized
    sum += e;
  }
  red[t] = sum; __syncthreads();
  for (int s = 128; s > 0; s >>= 1) { if (t < s) red[t] += red[t + s]; __syncthreads(); }
  const float inv = 1.0f / red[0];
  __syncthreads();
  for (int i = t; i < L_SZ; i += 256) {
    float w = w_ws[i] * inv;
    w_ws[i] = w;
    w_out[i] = w;
  }
}

// ---------------------------------------------------------------------------
// attn_applied[h] = sum_i w[i]*enc[i][h]  (column reduction, coalesced in h)
// also xcat[h] = emb[tok][h]; xcat[H+h] = attn_applied[h]
// ---------------------------------------------------------------------------
__global__ void attn_embed(const float* __restrict__ enc,
                           const float* __restrict__ attw,
                           const float* __restrict__ emb,
                           const int* __restrict__ tok,
                           float* __restrict__ xcat) {
  const int h = blockIdx.x * blockDim.x + threadIdx.x;   // 0..H-1
  float acc = 0.0f;
  for (int i = 0; i < L_SZ; i += 4) {
    acc += attw[i + 0] * enc[(size_t)(i + 0) * H_SZ + h];
    acc += attw[i + 1] * enc[(size_t)(i + 1) * H_SZ + h];
    acc += attw[i + 2] * enc[(size_t)(i + 2) * H_SZ + h];
    acc += attw[i + 3] * enc[(size_t)(i + 3) * H_SZ + h];
  }
  xcat[H_SZ + h] = acc;
  xcat[h] = emb[(size_t)tok[0] * H_SZ + h];
}

// ---------------------------------------------------------------------------
// GRU gates for one layer. g = [gx(3H) | gh(3H)] (biases already added).
// ---------------------------------------------------------------------------
__global__ void gru_gate(const float* __restrict__ g,
                         const float* __restrict__ h_old,
                         float* __restrict__ h_new,
                         float* __restrict__ x_buf) {
  const int j = blockIdx.x * blockDim.x + threadIdx.x;   // 0..H-1
  const float gxr = g[j],            ghr = g[3 * H_SZ + j];
  const float gxz = g[H_SZ + j],     ghz = g[4 * H_SZ + j];
  const float gxn = g[2 * H_SZ + j], ghn = g[5 * H_SZ + j];
  const float r = 1.0f / (1.0f + expf(-(gxr + ghr)));
  const float z = 1.0f / (1.0f + expf(-(gxz + ghz)));
  const float n = tanhf(gxn + r * ghn);
  const float hv = (1.0f - z) * n + z * h_old[j];
  h_new[j] = hv;
  x_buf[j] = hv;
}

// ---------------------------------------------------------------------------
// Log-sum-exp stats over the V logits (single block, two passes through L2).
// ---------------------------------------------------------------------------
__global__ void lse_stats(const float* __restrict__ logits, float* __restrict__ red2) {
  __shared__ float sm[256];
  const int t = threadIdx.x;
  float m = -INFINITY;
  for (int i = t; i < V_SZ; i += 256) m = fmaxf(m, logits[i]);
  sm[t] = m; __syncthreads();
  for (int s = 128; s > 0; s >>= 1) { if (t < s) sm[t] = fmaxf(sm[t], sm[t + s]); __syncthreads(); }
  const float mx = sm[0];
  __syncthreads();
  float sum = 0.0f;
  for (int i = t; i < V_SZ; i += 256) sum += expf(logits[i] - mx);
  sm[t] = sum; __syncthreads();
  for (int s = 128; s > 0; s >>= 1) { if (t < s) sm[t] += sm[t + s]; __syncthreads(); }
  if (t == 0) { red2[0] = mx; red2[1] = logf(sm[0]); }
}

__global__ void logsoftmax_final(const float* __restrict__ logits,
                                 const float* __restrict__ red2,
                                 float* __restrict__ out) {
  const int v = blockIdx.x * blockDim.x + threadIdx.x;
  if (v < V_SZ) out[v] = logits[v] - red2[0] - red2[1];
}

// ---------------------------------------------------------------------------
// Host-side orchestration
// ---------------------------------------------------------------------------
static inline int gemv_grid(int rows) { return (rows + 127) / 128; }  // 8 waves * 16 rows

extern "C" void kernel_launch(void* const* d_in, const int* in_sizes, int n_in,
                              void* d_out, int out_size, void* d_ws, size_t ws_size,
                              hipStream_t stream) {
  (void)in_sizes; (void)n_in; (void)out_size; (void)ws_size;

  const int*   input_token = (const int*)  d_in[0];
  const float* hidden      = (const float*)d_in[1];   // NL x 1 x H
  const float* enc         = (const float*)d_in[2];   // L x H
  const float* emb         = (const float*)d_in[3];   // V x H
  const float* attn2_w     = (const float*)d_in[4];   // 1 x 2H  ([w_h | w_e])
  // attn2_b (d_in[5]) and the h@w_h term are constant over the softmax axis ->
  // softmax is shift-invariant, so they are dropped (identical attn_weights).
  const float* comb_w      = (const float*)d_in[6];   // H x 2H
  const float* comb_b      = (const float*)d_in[7];   // H
  const float* gru_w_ih    = (const float*)d_in[8];   // NL x 3H x H
  const float* gru_w_hh    = (const float*)d_in[9];   // NL x 3H x H
  const float* gru_b_ih    = (const float*)d_in[10];  // NL x 3H
  const float* gru_b_hh    = (const float*)d_in[11];  // NL x 3H
  const float* out_w       = (const float*)d_in[12];  // V x H
  const float* out_b       = (const float*)d_in[13];  // V

  float* out_logits = (float*)d_out;                  // V
  float* out_hidden = out_logits + V_SZ;              // NL*H
  float* out_attnw  = out_hidden + NL_SZ * H_SZ;      // L

  // Workspace layout (floats)
  float* ws      = (float*)d_ws;
  float* scores  = ws;                    // L
  float* attw    = scores + L_SZ;         // L
  float* xcat    = attw + L_SZ;           // 2H
  float* xbuf    = xcat + 2 * H_SZ;       // H
  float* gbuf    = xbuf + H_SZ;           // 6H
  float* logits  = gbuf + 6 * H_SZ;       // V
  float* red2    = logits + V_SZ;         // 2

  // 1) scores = enc @ w_e   (w_e = attn2_w[H:])
  gemv_wmma_bf16<<<gemv_grid(L_SZ), 256, 0, stream>>>(
      enc, attn2_w + H_SZ, nullptr, scores, L_SZ, H_SZ, 0);

  // 2) attn_weights = softmax(scores)
  softmax2048<<<1, 256, 0, stream>>>(scores, attw, out_attnw);

  // 3) xcat = [emb[token], attn_weights @ enc]
  attn_embed<<<H_SZ / 256, 256, 0, stream>>>(enc, attw, emb, input_token, xcat);

  // 4) x = relu(comb_w @ xcat + comb_b)
  gemv_wmma_bf16<<<gemv_grid(H_SZ), 256, 0, stream>>>(
      comb_w, xcat, comb_b, xbuf, H_SZ, 2 * H_SZ, 1);

  // 5) GRU stack
  for (int l = 0; l < NL_SZ; ++l) {
    const float* w_ih = gru_w_ih + (size_t)l * 3 * H_SZ * H_SZ;
    const float* w_hh = gru_w_hh + (size_t)l * 3 * H_SZ * H_SZ;
    const float* b_ih = gru_b_ih + (size_t)l * 3 * H_SZ;
    const float* b_hh = gru_b_hh + (size_t)l * 3 * H_SZ;
    const float* h_l  = hidden + (size_t)l * H_SZ;

    gemv_wmma_bf16<<<gemv_grid(3 * H_SZ), 256, 0, stream>>>(
        w_ih, xbuf, b_ih, gbuf, 3 * H_SZ, H_SZ, 0);
    gemv_wmma_bf16<<<gemv_grid(3 * H_SZ), 256, 0, stream>>>(
        w_hh, h_l, b_hh, gbuf + 3 * H_SZ, 3 * H_SZ, H_SZ, 0);
    gru_gate<<<H_SZ / 256, 256, 0, stream>>>(
        gbuf, h_l, out_hidden + (size_t)l * H_SZ, xbuf);
  }

  // 6) logits = out_w @ x + out_b      (dominant: 206 MB streamed)
  gemv_wmma_bf16<<<gemv_grid(V_SZ), 256, 0, stream>>>(
      out_w, xbuf, out_b, logits, V_SZ, H_SZ, 0);

  // 7) log_softmax
  lse_stats<<<1, 256, 0, stream>>>(logits, red2);
  logsoftmax_final<<<(V_SZ + 255) / 256, 256, 0, stream>>>(logits, red2, out_logits);
}
